// Biological_module_84181359001849
// MI455X (gfx1250) — matrix operand order repaired and checked by reading, not compiled, not run on hardware
//
#include <hip/hip_runtime.h>
#include <math.h>

// ---------------------------------------------------------------------------
// Problem constants (match reference)
// ---------------------------------------------------------------------------
#define INPUT_DIM 20000   // K
#define UNITS     4096    // N
#define BATCH     2048    // M
#define BK        32      // K-chunk per WMMA step

typedef _Float16 v16h __attribute__((ext_vector_type(16)));
typedef _Float16 v4h  __attribute__((ext_vector_type(4)));
typedef float    v8f  __attribute__((ext_vector_type(8)));

// Workspace layout (bytes):
//   [0)            denseF32 : fp32 W^T  [UNITS][INPUT_DIM]  327,680,000
//   [327,680,000)  denseF16 : f16  W^T  [UNITS][INPUT_DIM]  163,840,000
//   [491,520,000)  xF16     : f16  x    [BATCH][INPUT_DIM]   81,920,000
#define WS_OFF_WF16 327680000L
#define WS_OFF_XF16 491520000L

// ---------------------------------------------------------------------------
// 1) zero the fp32 accumulation buffer
// ---------------------------------------------------------------------------
__global__ void zero_f32(float* __restrict__ p, long n4) {
  long i = (long)blockIdx.x * blockDim.x + threadIdx.x;
  long stride = (long)gridDim.x * blockDim.x;
  float4 z = make_float4(0.f, 0.f, 0.f, 0.f);
  for (; i < n4; i += stride) reinterpret_cast<float4*>(p)[i] = z;
}

// ---------------------------------------------------------------------------
// 2) COO scatter-add into W^T (fp32 atomics => exact duplicate handling)
// ---------------------------------------------------------------------------
__global__ void scatter_coo(const float* __restrict__ vals,
                            const int* __restrict__ ind,
                            float* __restrict__ dense, int nnz) {
  int i = blockIdx.x * blockDim.x + threadIdx.x;
  if (i >= nnz) return;
  int r = ind[2 * i];       // K index
  int c = ind[2 * i + 1];   // N index
  atomicAdd(dense + (long)c * INPUT_DIM + r, vals[i]);
}

// ---------------------------------------------------------------------------
// 3) fp32 -> f16 (vectorized 4-wide grid-stride)
// ---------------------------------------------------------------------------
__global__ void f32_to_f16(const float* __restrict__ src,
                           _Float16* __restrict__ dst, long n4) {
  long i = (long)blockIdx.x * blockDim.x + threadIdx.x;
  long stride = (long)gridDim.x * blockDim.x;
  for (; i < n4; i += stride) {
    float4 v = reinterpret_cast<const float4*>(src)[i];
    v4h h;
    h.x = (_Float16)v.x; h.y = (_Float16)v.y;
    h.z = (_Float16)v.z; h.w = (_Float16)v.w;
    reinterpret_cast<v4h*>(dst)[i] = h;
  }
}

// ---------------------------------------------------------------------------
// CDNA5 async global -> LDS copy (16B per lane, per-lane LDS dest address).
// Tracked by ASYNCcnt; completion published via s_wait_asynccnt + barrier.
// ---------------------------------------------------------------------------
__device__ __forceinline__ void async_copy_b128(unsigned lds_byte_addr,
                                                const void* gaddr) {
  asm volatile("global_load_async_to_lds_b128 %0, %1, off"
               :: "v"(lds_byte_addr), "v"(gaddr)
               : "memory");
}
__device__ __forceinline__ void wait_asynccnt0() {
  asm volatile("s_wait_asynccnt 0x0" ::: "memory");
}

// ---------------------------------------------------------------------------
// 4) WMMA GEMM: out = tanh(x @ W + bias)
//    A   = xF16  [BATCH][INPUT_DIM]   (M x K, K contiguous)
//    Bt  = WF16  [UNITS][INPUT_DIM]   (N x K, K contiguous)
//    out = fp32  [BATCH][UNITS]
//
//    Block tile 128x128, 8 wave32, wave tile 32(M) x 64(N) => 8 WMMAs per
//    K=32 step, 625 steps. Double-buffered LDS; next tile is DMA'd
//    global->LDS with GLOBAL_LOAD_ASYNC_TO_LDS_B128 while WMMAs run.
//
//    LDS holds tiles pre-permuted into the exact WMMA fragment layout
//    (ISA 7.12.2), so a lane's v16h fragment is 32 contiguous bytes:
//      A frag (16x32): lane group = (k/8)&1, h = k%8 + 8*(k/16)
//      B frag (32x16): lane group = (k/8)&1, h = k%8 + 8*(k/16)
// ---------------------------------------------------------------------------
__global__ __launch_bounds__(256)
void gemm_wmma_tanh(const _Float16* __restrict__ A,
                    const _Float16* __restrict__ Bt,
                    const float* __restrict__ bias,
                    float* __restrict__ out) {
  __shared__ __align__(32) _Float16 As[2][8 * 32 * 16];  // [buf][mt*512 + lane*16 + h]
  __shared__ __align__(32) _Float16 Bs[2][8 * 32 * 16];  // [buf][nt*512 + lane*16 + h]
  const unsigned BUF_BYTES = 8 * 32 * 16 * 2;            // 8 KB per buffer

  const int t    = threadIdx.x;
  const int lane = t & 31;
  const int wave = t >> 5;
  const int wm   = wave >> 1;  // 0..3 : wave row (32 rows)
  const int wn   = wave & 1;   // 0..1 : wave col (64 cols)
  const int row0 = blockIdx.y * 128;
  const int col0 = blockIdx.x * 128;

  // --- staging: 512 b128 chunks per operand tile, 2 per thread ------------
  // chunk q: r = q>>2 (tile row 0..127), c = q&3 (which 8 K-halves)
  long gA[2], gB[2];
  unsigned lAb[2], lBb[2];   // LDS byte addresses in buffer 0
#pragma unroll
  for (int j = 0; j < 2; ++j) {
    int q = t + 256 * j;
    int r = q >> 2, c = q & 3;
    gA[j] = (long)(row0 + r) * INPUT_DIM + c * 8;
    gB[j] = (long)(col0 + r) * INPUT_DIM + c * 8;
    int hA = (r >> 4) * 512 + (((r & 15) + 16 * (c & 1)) * 16) + 8 * (c >> 1);
    int hB = (r >> 4) * 512 + (((r & 15) + 16 * (c & 1)) * 16) + 8 * (c >> 1);
    lAb[j] = (unsigned)(uintptr_t)(&As[0][hA]);  // low 32 bits == LDS offset
    lBb[j] = (unsigned)(uintptr_t)(&Bs[0][hB]);
  }

  // --- prologue: DMA K-chunk 0 into buffer 0 ------------------------------
  async_copy_b128(lAb[0], A  + gA[0]);
  async_copy_b128(lAb[1], A  + gA[1]);
  async_copy_b128(lBb[0], Bt + gB[0]);
  async_copy_b128(lBb[1], Bt + gB[1]);
  wait_asynccnt0();
  __syncthreads();

  v8f acc[2][4];
#pragma unroll
  for (int i = 0; i < 2; ++i)
#pragma unroll
    for (int j = 0; j < 4; ++j) {
      v8f z = {0.f, 0.f, 0.f, 0.f, 0.f, 0.f, 0.f, 0.f};
      acc[i][j] = z;
    }

  const int NK = INPUT_DIM / BK;  // 625
  int p = 0;
  for (int kb = 0; kb < NK; ++kb) {
    const bool more = (kb + 1 < NK);
    if (more) {
      // DMA next K-chunk into the other buffer; overlaps with WMMAs below.
      const long ko = (long)(kb + 1) * BK;
      const unsigned boff = (unsigned)(p ^ 1) * BUF_BYTES;
      async_copy_b128(lAb[0] + boff, A  + gA[0] + ko);
      async_copy_b128(lAb[1] + boff, A  + gA[1] + ko);
      async_copy_b128(lBb[0] + boff, Bt + gB[0] + ko);
      async_copy_b128(lBb[1] + boff, Bt + gB[1] + ko);
    }

    v16h af[2], bf[4];
#pragma unroll
    for (int i = 0; i < 2; ++i)
      af[i] = *reinterpret_cast<const v16h*>(&As[p][(wm * 2 + i) * 512 + lane * 16]);
#pragma unroll
    for (int j = 0; j < 4; ++j)
      bf[j] = *reinterpret_cast<const v16h*>(&Bs[p][(wn * 4 + j) * 512 + lane * 16]);

#pragma unroll
    for (int i = 0; i < 2; ++i)
#pragma unroll
      for (int j = 0; j < 4; ++j)
        acc[i][j] = __builtin_amdgcn_wmma_f32_16x16x32_f16(
            /*neg_a=*/false, af[i], /*neg_b=*/false, bf[j],
            /*c_mod=*/(short)0, acc[i][j],
            /*reuse_a=*/false, /*reuse_b=*/false);

    if (more) wait_asynccnt0();   // next buffer fully landed in LDS
    __syncthreads();              // ... and visible to all waves
    p ^= 1;
  }

  // --- epilogue: bias + tanh, C layout per ISA (lane = col, VGPR = row) ---
  const int cLane = lane & 15;
  const int rHalf = (lane >> 4) * 8;
  const int colBase = col0 + wn * 64;
  const int rowBase = row0 + wm * 32;
#pragma unroll
  for (int i = 0; i < 2; ++i) {
#pragma unroll
    for (int j = 0; j < 4; ++j) {
      const int ccol = colBase + j * 16 + cLane;
      const float bv = bias[ccol];
#pragma unroll
      for (int r = 0; r < 8; ++r) {
        const int crow = rowBase + i * 16 + rHalf + r;
        __builtin_nontemporal_store(tanhf(acc[i][j][r] + bv),
                                    out + (long)crow * UNITS + ccol);
      }
    }
  }
}

// ---------------------------------------------------------------------------
// Launch: inputs are {x (f32), kernel_vector (f32), bias (f32), nonzero_ind (i32)}
// ---------------------------------------------------------------------------
extern "C" void kernel_launch(void* const* d_in, const int* in_sizes, int n_in,
                              void* d_out, int out_size, void* d_ws, size_t ws_size,
                              hipStream_t stream) {
  const float* x    = (const float*)d_in[0];
  const float* kv   = (const float*)d_in[1];
  const float* bias = (const float*)d_in[2];
  const int*   ind  = (const int*)d_in[3];
  float*       out  = (float*)d_out;
  const int    nnz  = in_sizes[1];

  char* ws = (char*)d_ws;
  float*    denseF32 = (float*)ws;
  _Float16* denseF16 = (_Float16*)(ws + WS_OFF_WF16);
  _Float16* xF16     = (_Float16*)(ws + WS_OFF_XF16);

  const long wElems = (long)UNITS * INPUT_DIM;   // 81,920,000
  const long xElems = (long)BATCH * INPUT_DIM;   // 40,960,000

  zero_f32<<<2048, 256, 0, stream>>>(denseF32, wElems / 4);
  scatter_coo<<<(nnz + 255) / 256, 256, 0, stream>>>(kv, ind, denseF32, nnz);
  f32_to_f16<<<2048, 256, 0, stream>>>(denseF32, denseF16, wElems / 4);
  f32_to_f16<<<1024, 256, 0, stream>>>(x, xF16, xElems / 4);

  dim3 grid(UNITS / 128, BATCH / 128);  // (32, 16)
  gemm_wmma_tanh<<<grid, 256, 0, stream>>>(xF16, denseF16, bias, out);
}
